// VQ_VAE_88673894793401
// MI455X (gfx1250) — matrix operand order repaired and checked by reading, not compiled, not run on hardware
//
#include <hip/hip_runtime.h>
#include <hip/hip_bf16.h>
#include <stdint.h>

typedef __attribute__((ext_vector_type(16))) _Float16 v16h;
typedef __attribute__((ext_vector_type(8)))  _Float16 v8h;
typedef __attribute__((ext_vector_type(8)))  float    v8f;
typedef __attribute__((ext_vector_type(4)))  unsigned int v4u;
typedef __attribute__((ext_vector_type(4)))  int v4i;
typedef __attribute__((ext_vector_type(8)))  int v8i;

#define TB 128   // 4 waves of 32 (wave32 on gfx1250)

#if defined(__has_builtin)
#if __has_builtin(__builtin_amdgcn_tensor_load_to_lds)
#define HAVE_TDM 1
#endif
#endif

// ---------------------------------------------------------------- utilities

__global__ void k_zero_f32(float* p, int n) {
  int i = blockIdx.x * blockDim.x + threadIdx.x;
  if (i < n) p[i] = 0.0f;
}

// f32 NCHW -> f16 NHWC
__global__ void k_nchw_to_nhwc_h(const float* __restrict__ x, _Float16* __restrict__ o,
                                 int N, int C, int H, int W) {
  long long idx = (long long)blockIdx.x * blockDim.x + threadIdx.x;
  long long tot = (long long)N * C * H * W;
  if (idx >= tot) return;
  int c = (int)(idx % C); long long t = idx / C;
  int w = (int)(t % W); t /= W;
  int h = (int)(t % H); int n = (int)(t / H);
  o[idx] = (_Float16)x[(((long long)n * C + c) * H + h) * W + w];
}

// Pack weights to f16 [Kpad x Npad], K order = (kh*KW + kw)*Cin + ci, zero padded.
// transposed==0: w is OIHW (Cout,Cin,KH,KW).  transposed==1: w is (Cin,Cout,KH,KW).
__global__ void k_pack_w(const float* __restrict__ w, _Float16* __restrict__ o,
                         int Cin, int Cout, int KH, int KW, int Kpad, int Npad,
                         int transposed) {
  int idx = blockIdx.x * blockDim.x + threadIdx.x;
  if (idx >= Kpad * Npad) return;
  int k = idx / Npad, n = idx % Npad;
  float v = 0.0f;
  int K = Cin * KH * KW;
  if (k < K && n < Cout) {
    int ci = k % Cin; int t = k / Cin;
    int kwi = t % KW; int khi = t / KW;
    long long src = transposed
        ? ((((long long)ci * Cout + n) * KH + khi) * KW + kwi)
        : ((((long long)n * Cin + ci) * KH + khi) * KW + kwi);
    v = w[src];
  }
  o[(long long)k * Npad + n] = (_Float16)v;
}

// ------------------------------------------------ implicit-GEMM conv (WMMA)
// MODE: 0 = conv (stride,pad) ; 1 = transposed conv ; 2 = plain GEMM (1x1, K=Cin)
// VEC : Cin % 16 == 0 -> a thread's 16-half K-window sits inside ONE (kh,kw)
//       tap, so the A gather is two aligned b128 global loads + one check.
// Block = 128 threads = 4 waves; block computes a 64(M) x NT*16(N) tile.
// B chunk (32 x Npad) DMA'd into LDS by the Tensor Data Mover (TENSORcnt
// fenced), with a vectorized cooperative-copy fallback.
template<int NT, int MODE, bool VEC>
__global__ __launch_bounds__(TB)
void k_conv_wmma(const _Float16* __restrict__ in, const _Float16* __restrict__ Bp,
                 const float* __restrict__ bias, const _Float16* __restrict__ resid,
                 _Float16* __restrict__ out16, float* __restrict__ out32nchw,
                 int M, int K, int Cin, int Cout,
                 int Hin, int Win, int Ho, int Wo,
                 int KH, int KW, int stride, int pad, int doRelu) {
  constexpr int NPAD = NT * 16;
  __shared__ __align__(32) _Float16 sA[64 * 32];
  __shared__ __align__(32) _Float16 sBraw[32 * NPAD];   // row-major [k][n]

  const int tid  = threadIdx.x;
  const int lane = tid & 31;
  const int wave = tid >> 5;
  const int mBase = blockIdx.x * 64;
  const int kChunks = (K + 31) >> 5;
  const int HoWo = Ho * Wo;

  // -------- per-thread A-row decomposition (hoisted out of the K loop) ----
  const int ml    = tid >> 1;          // A-tile row owned by this thread
  const int kHalf = (tid & 1) << 4;    // 0 or 16 within the 32-wide chunk
  const int mGlob = mBase + ml;
  const bool mValid = mGlob < M;
  int ihB = 0, iwB = 0;                // spatial bases (MODE 0/1)
  const _Float16* inN = in;
  if (MODE != 2) {
    int mc = mValid ? mGlob : 0;
    int n  = mc / HoWo; int rem = mc - n * HoWo;
    int oh = rem / Wo;  int ow  = rem - oh * Wo;
    inN = in + (long long)n * Hin * Win * Cin;
    if (MODE == 0) { ihB = oh * stride - pad; iwB = ow * stride - pad; }
    else           { ihB = oh + pad;          iwB = ow + pad; }
  }

  v8f acc[NT];
#pragma unroll
  for (int i = 0; i < NT; ++i) acc[i] = (v8f){0, 0, 0, 0, 0, 0, 0, 0};

  for (int kc = 0; kc < kChunks; ++kc) {
    // ---- B chunk: Tensor Data Mover DMA (global -> LDS) ----
#ifdef HAVE_TDM
    if (wave == 0) {
      unsigned ldsOff = (unsigned)(size_t)&sBraw[0];
      unsigned long long ga =
          (unsigned long long)(size_t)(Bp + (size_t)(kc << 5) * NPAD);
      v4u g0;
      g0.x = 1u;                                     // count=1 (valid D#)
      g0.y = ldsOff;                                 // lds_addr
      g0.z = (unsigned)(ga & 0xFFFFFFFFu);           // global_addr[31:0]
      g0.w = (unsigned)((ga >> 32) & 0x01FFFFFFu) | (2u << 30);  // [56:32]|type=2
      v8i g1;
      g1[0] = 1 << 16;                               // data_size = 2 bytes
      g1[1] = NPAD << 16;                            // tensor_dim0[15:0]
      g1[2] = 32 << 16;                              // tensor_dim1[15:0] = 32 rows
      g1[3] = NPAD << 16;                            // tile_dim0 = NPAD
      g1[4] = 32;                                    // tile_dim1 = 32 rows
      g1[5] = NPAD;                                  // tensor_dim0_stride
      g1[6] = 0;
      g1[7] = 0;
      v4i z4 = (v4i){0, 0, 0, 0};
#if __clang_major__ >= 23
      v8i z8 = (v8i){0, 0, 0, 0, 0, 0, 0, 0};
      __builtin_amdgcn_tensor_load_to_lds(g0, g1, z4, z4, z8, 0);
#else
      __builtin_amdgcn_tensor_load_to_lds(g0, g1, z4, z4, 0);
#endif
    }
#else
    // fallback: vectorized cooperative copy (b128 per thread-step)
    for (int idx = tid; idx < NPAD * 4; idx += TB) {
      int off = idx * 8;                 // half index within 32 x NPAD tile
      int row = off / NPAD;              // NPAD is compile-time -> shifts
      int cn  = off - row * NPAD;
      *(v8h*)&sBraw[off] =
          *(const v8h*)&Bp[((long long)((kc << 5) + row)) * NPAD + cn];
    }
    if (kc + 1 < kChunks && tid < 32)    // lowers to global_prefetch_b8
      __builtin_prefetch(&Bp[((long long)(((kc + 1) << 5)) + tid) * NPAD], 0, 1);
#endif

    // ---- A tile: fused im2col gather ----
    {
      const int k0 = (kc << 5) + kHalf;
      const bool aValid = mValid && (k0 < K);   // K is a multiple of 16
      v8h t0 = (v8h)(_Float16)0.0f, t1 = (v8h)(_Float16)0.0f;
      if (MODE == 2) {
        if (aValid) {
          const _Float16* p = in + (long long)mGlob * Cin + k0;
          t0 = *(const v8h*)p;
          t1 = *(const v8h*)(p + 8);
        }
      } else if (VEC) {
        // Cin % 16 == 0: whole window shares one (khi,kwi) tap
        int ci = k0 % Cin; int t = k0 / Cin;
        int kwi = t % KW;  int khi = t / KW;
        const _Float16* p = nullptr;
        if (aValid) {
          if (MODE == 0) {
            int ih = ihB + khi, iw = iwB + kwi;
            if (ih >= 0 && ih < Hin && iw >= 0 && iw < Win)
              p = &inN[((long long)ih * Win + iw) * Cin + ci];
          } else {
            int ihn = ihB - khi, iwn = iwB - kwi;
            if (ihn >= 0 && iwn >= 0 && (ihn % stride) == 0 && (iwn % stride) == 0) {
              int ih = ihn / stride, iw = iwn / stride;
              if (ih < Hin && iw < Win)
                p = &inN[((long long)ih * Win + iw) * Cin + ci];
            }
          }
        }
        if (p) {                        // two aligned global_load_b128
          t0 = *(const v8h*)p;
          t1 = *(const v8h*)(p + 8);
        }
      } else {
        // scalar path (e1: Cin == 3), carry-increment k -> (ci,kwi,khi)
        int ci = k0 % Cin; int t = k0 / Cin;
        int kwi = t % KW;  int khi = t / KW;
#pragma unroll
        for (int e = 0; e < 16; ++e) {
          _Float16 v = (_Float16)0.0f;
          if (aValid) {
            if (MODE == 0) {
              int ih = ihB + khi, iw = iwB + kwi;
              if (ih >= 0 && ih < Hin && iw >= 0 && iw < Win)
                v = inN[((long long)ih * Win + iw) * Cin + ci];
            } else {
              int ihn = ihB - khi, iwn = iwB - kwi;
              if (ihn >= 0 && iwn >= 0 && (ihn % stride) == 0 && (iwn % stride) == 0) {
                int ih = ihn / stride, iw = iwn / stride;
                if (ih < Hin && iw < Win)
                  v = inN[((long long)ih * Win + iw) * Cin + ci];
              }
            }
          }
          if (e < 8) t0[e] = v; else t1[e - 8] = v;
          if (++ci == Cin) { ci = 0; if (++kwi == KW) { kwi = 0; ++khi; } }
        }
      }
      *(v8h*)&sA[ml * 32 + kHalf]     = t0;   // ds_store_b128
      *(v8h*)&sA[ml * 32 + kHalf + 8] = t1;   // ds_store_b128
    }

#ifdef HAVE_TDM
    if (wave == 0) __builtin_amdgcn_s_wait_tensorcnt(0);
#endif
    __syncthreads();

    // ---- WMMA ----
    const int mf = wave * 16 + (lane & 15);
    const int hi = lane >> 4;
    v8h a0 = *(const v8h*)&sA[mf * 32 + hi * 8];        // K = 8*hi ..
    v8h a1 = *(const v8h*)&sA[mf * 32 + 16 + hi * 8];   // K = 16+8*hi ..
    v16h a = __builtin_shufflevector(a0, a1, 0, 1, 2, 3, 4, 5, 6, 7,
                                     8, 9, 10, 11, 12, 13, 14, 15);
    const int col = lane & 15;
#pragma unroll
    for (int nt = 0; nt < NT; ++nt) {
      v16h b;
#pragma unroll
      for (int j = 0; j < 16; ++j)                       // frag: K=(j&7)+8*hi+16*(j>>3), N=col
        b[j] = sBraw[((j & 7) + (hi << 3) + ((j >> 3) << 4)) * NPAD + nt * 16 + col];
      acc[nt] = __builtin_amdgcn_wmma_f32_16x16x32_f16(
          false, a, false, b, (short)0, acc[nt], false, false);
    }
    __syncthreads();
  }

  // ---- epilogue: bias (+residual) (+relu), f16 NHWC and/or f32 NCHW ----
  const int hiL  = lane >> 4;
  const int mRow = mBase + wave * 16;
#pragma unroll
  for (int nt = 0; nt < NT; ++nt) {
    int coln = nt * 16 + (lane & 15);
    if (coln >= Cout) continue;
#pragma unroll
    for (int r = 0; r < 8; ++r) {
      int mm = mRow + r + hiL * 8;   // C/D layout: M = r + 8*(lane>>4)
      if (mm >= M) continue;
      float v = acc[nt][r] + bias[coln];
      if (resid) v += (float)resid[(long long)mm * Cout + coln];
      if (doRelu) v = v > 0.0f ? v : 0.0f;
      if (out16) out16[(long long)mm * Cout + coln] = (_Float16)v;
      if (out32nchw) {
        int n = mm / HoWo; int rem = mm - n * HoWo;
        out32nchw[(((long long)n * Cout + coln) * Ho + rem / Wo) * Wo + (rem % Wo)] = v;
      }
    }
  }
}

// ------------------------------------------------------------ VQ quantizer
__global__ __launch_bounds__(256)
void k_vq(const _Float16* __restrict__ z, const float* __restrict__ emb,
          float* __restrict__ zq_nchw, _Float16* __restrict__ zq_h,
          float* __restrict__ lossAcc, int M, int NE, int Hq, int Wq) {
  extern __shared__ float semb[];   // 256*64 f32 = 64 KB
  for (int i = threadIdx.x; i < NE * 64; i += blockDim.x) semb[i] = emb[i];
  __syncthreads();
  int m = blockIdx.x * blockDim.x + threadIdx.x;
  if (m >= M) return;
  float zr[64];
#pragma unroll
  for (int c = 0; c < 64; ++c) zr[c] = (float)z[(long long)m * 64 + c];
  int best = 0; float bestd = 3.4e38f;
  for (int e = 0; e < NE; ++e) {
    const float* ep = &semb[e * 64];
    float d = 0.0f;
#pragma unroll
    for (int c = 0; c < 64; ++c) { float t = zr[c] - ep[c]; d = fmaf(t, t, d); }
    if (d < bestd) { bestd = d; best = e; }
  }
  int n = m / (Hq * Wq); int rem = m % (Hq * Wq);
  int oh = rem / Wq;     int ow = rem % Wq;
  const float* ep = &semb[best * 64];
#pragma unroll
  for (int c = 0; c < 64; ++c) {
    float q = ep[c];
    zq_h[(long long)m * 64 + c] = (_Float16)q;                    // decoder input
    zq_nchw[((((long long)n * 64) + c) * Hq + oh) * Wq + ow] = q; // output tuple[1]
  }
  atomicAdd(lossAcc, bestd);   // sum of ||z - e_best||^2
}

__global__ void k_loss_final(const float* __restrict__ acc, float* __restrict__ out,
                             float invCount) {
  if (blockIdx.x == 0 && threadIdx.x == 0)
    out[0] = 1.25f * acc[0] * invCount;   // q_loss + 0.25*e_loss
}

// ------------------------------------------------------------------- launch

extern "C" void kernel_launch(void* const* d_in, const int* in_sizes, int n_in,
                              void* d_out, int out_size, void* d_ws, size_t ws_size,
                              hipStream_t stream) {
  (void)in_sizes; (void)n_in; (void)out_size; (void)ws_size;

  const float* x    = (const float*)d_in[0];
  const float* e1w  = (const float*)d_in[1];  const float* e1b  = (const float*)d_in[2];
  const float* e2w  = (const float*)d_in[3];  const float* e2b  = (const float*)d_in[4];
  const float* e3w  = (const float*)d_in[5];  const float* e3b  = (const float*)d_in[6];
  const float* r1w1 = (const float*)d_in[7];  const float* r1b1 = (const float*)d_in[8];
  const float* r1w2 = (const float*)d_in[9];  const float* r1b2 = (const float*)d_in[10];
  const float* r2w1 = (const float*)d_in[11]; const float* r2b1 = (const float*)d_in[12];
  const float* r2w2 = (const float*)d_in[13]; const float* r2b2 = (const float*)d_in[14];
  const float* s1w1 = (const float*)d_in[15]; const float* s1b1 = (const float*)d_in[16];
  const float* s1w2 = (const float*)d_in[17]; const float* s1b2 = (const float*)d_in[18];
  const float* s2w1 = (const float*)d_in[19]; const float* s2b1 = (const float*)d_in[20];
  const float* s2w2 = (const float*)d_in[21]; const float* s2b2 = (const float*)d_in[22];
  const float* d1w  = (const float*)d_in[23]; const float* d1b  = (const float*)d_in[24];
  const float* d2w  = (const float*)d_in[25]; const float* d2b  = (const float*)d_in[26];
  const float* d3w  = (const float*)d_in[27]; const float* d3b  = (const float*)d_in[28];
  const float* emb  = (const float*)d_in[29];

  float* out = (float*)d_out;
  const int NB = 128;
  const int H3 = 132;                                   // (64-1)*2 - 2 + 8
  const long long outElems = (long long)NB * 3 * H3 * H3;
  const long long zqOff    = outElems;
  const long long lossOff  = outElems + (long long)NB * 64 * 16 * 16;

  char* wsb = (char*)d_ws;
  size_t off = 0;
  auto alloc = [&](size_t bytes) -> void* {
    size_t a = (off + 255) & ~(size_t)255; off = a + bytes; return (void*)(wsb + a);
  };
  auto allocH = [&](long long n) -> _Float16* { return (_Float16*)alloc((size_t)n * 2); };

  _Float16* xh   = allocH(6291456);
  _Float16* aE1  = allocH((long long)524288 * 32);
  _Float16* aE2  = allocH((long long)131072 * 64);
  _Float16* bZ   = allocH((long long)32768 * 64);
  _Float16* bT   = allocH((long long)32768 * 64);
  _Float16* bR1  = allocH((long long)32768 * 64);
  _Float16* bR2  = allocH((long long)32768 * 64);
  _Float16* zqH  = allocH((long long)32768 * 64);
  _Float16* bS1  = allocH((long long)32768 * 64);
  _Float16* bS2  = allocH((long long)32768 * 64);
  _Float16* bD1  = allocH((long long)131072 * 64);
  _Float16* bD2  = allocH((long long)524288 * 32);
  _Float16* pwE1 = allocH(64 * 32);
  _Float16* pwE2 = allocH(512 * 64);
  _Float16* pwE3 = allocH(1024 * 64);
  _Float16* pwR1a = allocH(576 * 64), *pwR1b = allocH(64 * 64);
  _Float16* pwR2a = allocH(576 * 64), *pwR2b = allocH(64 * 64);
  _Float16* pwS1a = allocH(576 * 64), *pwS1b = allocH(64 * 64);
  _Float16* pwS2a = allocH(576 * 64), *pwS2b = allocH(64 * 64);
  _Float16* pwD1 = allocH(1024 * 64);
  _Float16* pwD2 = allocH(1024 * 32);
  _Float16* pwD3 = allocH(2048 * 16);
  float* lossAcc = (float*)alloc(sizeof(float));

  dim3 blk(TB);
  k_zero_f32<<<1, 32, 0, stream>>>(lossAcc, 1);
  {
    long long tot = 6291456;
    k_nchw_to_nhwc_h<<<dim3((unsigned)((tot + 255) / 256)), 256, 0, stream>>>(
        x, xh, NB, 3, 128, 128);
  }
  auto pack = [&](const float* w, _Float16* o, int Cin, int Cout, int KH, int KW,
                  int Kpad, int Npad, int tr) {
    int n = Kpad * Npad;
    k_pack_w<<<dim3((n + 255) / 256), 256, 0, stream>>>(w, o, Cin, Cout, KH, KW,
                                                        Kpad, Npad, tr);
  };
  pack(e1w, pwE1, 3, 32, 4, 4, 64, 32, 0);
  pack(e2w, pwE2, 32, 64, 4, 4, 512, 64, 0);
  pack(e3w, pwE3, 64, 64, 4, 4, 1024, 64, 0);
  pack(r1w1, pwR1a, 64, 64, 3, 3, 576, 64, 0);  pack(r1w2, pwR1b, 64, 64, 1, 1, 64, 64, 0);
  pack(r2w1, pwR2a, 64, 64, 3, 3, 576, 64, 0);  pack(r2w2, pwR2b, 64, 64, 1, 1, 64, 64, 0);
  pack(s1w1, pwS1a, 64, 64, 3, 3, 576, 64, 0);  pack(s1w2, pwS1b, 64, 64, 1, 1, 64, 64, 0);
  pack(s2w1, pwS2a, 64, 64, 3, 3, 576, 64, 0);  pack(s2w2, pwS2b, 64, 64, 1, 1, 64, 64, 0);
  pack(d1w, pwD1, 64, 64, 4, 4, 1024, 64, 1);
  pack(d2w, pwD2, 64, 32, 4, 4, 1024, 32, 1);
  pack(d3w, pwD3, 32, 3, 8, 8, 2048, 16, 1);

  { int M = 524288;
    k_conv_wmma<2, 0, false><<<dim3((M + 63) / 64), blk, 0, stream>>>(
        xh, pwE1, e1b, nullptr, aE1, nullptr,
        M, 48, 3, 32, 128, 128, 64, 64, 4, 4, 2, 1, 1); }
  { int M = 131072;
    k_conv_wmma<4, 0, true><<<dim3((M + 63) / 64), blk, 0, stream>>>(
        aE1, pwE2, e2b, nullptr, aE2, nullptr,
        M, 512, 32, 64, 64, 64, 32, 32, 4, 4, 2, 1, 1); }
  { int M = 32768;
    k_conv_wmma<4, 0, true><<<dim3((M + 63) / 64), blk, 0, stream>>>(
        aE2, pwE3, e3b, nullptr, bZ, nullptr,
        M, 1024, 64, 64, 32, 32, 16, 16, 4, 4, 2, 1, 0); }

  const int Mq = 32768;
  auto resblock = [&](const _Float16* xin, _Float16* tmp, _Float16* o,
                      const _Float16* wa, const float* ba,
                      const _Float16* wb, const float* bb) {
    k_conv_wmma<4, 0, true><<<dim3((Mq + 63) / 64), blk, 0, stream>>>(
        xin, wa, ba, nullptr, tmp, nullptr,
        Mq, 576, 64, 64, 16, 16, 16, 16, 3, 3, 1, 1, 1);
    k_conv_wmma<4, 2, true><<<dim3((Mq + 63) / 64), blk, 0, stream>>>(
        tmp, wb, bb, xin, o, nullptr,
        Mq, 64, 64, 64, 16, 16, 16, 16, 1, 1, 1, 0, 1);
  };
  resblock(bZ,  bT, bR1, pwR1a, r1b1, pwR1b, r1b2);
  resblock(bR1, bT, bR2, pwR2a, r2b1, pwR2b, r2b2);

  k_vq<<<dim3(Mq / 256), 256, 256 * 64 * sizeof(float), stream>>>(
      bR2, emb, out + zqOff, zqH, lossAcc, Mq, 256, 16, 16);

  resblock(zqH, bT, bS1, pwS1a, s1b1, pwS1b, s1b2);
  resblock(bS1, bT, bS2, pwS2a, s2b1, pwS2b, s2b2);

  { int M = 131072;
    k_conv_wmma<4, 1, true><<<dim3((M + 63) / 64), blk, 0, stream>>>(
        bS2, pwD1, d1b, nullptr, bD1, nullptr,
        M, 1024, 64, 64, 16, 16, 32, 32, 4, 4, 2, 1, 1); }
  { int M = 524288;
    k_conv_wmma<2, 1, true><<<dim3((M + 63) / 64), blk, 0, stream>>>(
        bD1, pwD2, d2b, nullptr, bD2, nullptr,
        M, 1024, 64, 32, 32, 32, 64, 64, 4, 4, 2, 1, 1); }
  { int M = NB * H3 * H3;
    k_conv_wmma<1, 1, true><<<dim3((M + 63) / 64), blk, 0, stream>>>(
        bD2, pwD3, d3b, nullptr, nullptr, out,
        M, 2048, 32, 3, 64, 64, H3, H3, 8, 8, 2, 1, 0); }

  k_loss_final<<<1, 32, 0, stream>>>(lossAcc, out + lossOff,
                                     1.0f / (float)(Mq * 64));
}
